// LocalAttention_19439021981698
// MI455X (gfx1250) — compile-verified, matrix operand order checked by reading, not run on hardware
//
#include <hip/hip_runtime.h>
#include <hip/hip_bf16.h>
#include <math.h>

#define N_TOK   32768
#define DIM     1024
#define HEADS   8
#define DH      64
#define INNER   512
#define QKV_N   1536
#define WS      256
#define NWIN    128
#define KSTRIDE ((N_TOK + WS) * DH)     // per-head K stride incl. guard window
#define VROW    (N_TOK + WS)            // per-d V row stride incl. guard window
#define MASK_VALF (-1.0e10f)
#define LN_EPS  1e-5f

typedef __bf16 bf16_t;
typedef __attribute__((ext_vector_type(8)))  __bf16 v8bf;
typedef __attribute__((ext_vector_type(16))) __bf16 v16bf;
typedef __attribute__((ext_vector_type(8)))  float  v8f;

#define WMMA_BF16(A, B, C) \
  __builtin_amdgcn_wmma_f32_16x16x32_bf16(false, (A), false, (B), (short)0, (C), false, false)

__device__ __forceinline__ v16bf combine(v8bf lo, v8bf hi) {
  return __builtin_shufflevector(lo, hi, 0, 1, 2, 3, 4, 5, 6, 7,
                                         8, 9, 10, 11, 12, 13, 14, 15);
}

// A-style fragment: lane's 16 values are at p[0..7] and p[16..23].
__device__ __forceinline__ v16bf ldfrag(const bf16_t* p) {
  return combine(*(const v8bf*)p, *(const v8bf*)(p + 16));
}
// Packed-B fragment: lane's 16 values are 32 contiguous bytes.
__device__ __forceinline__ v16bf ldfrag32(const bf16_t* p) {
  return *(const v16bf*)p;
}

// Copy-free ping-pong helpers for the dense GEMMs: each LOADSET overwrites a
// full fragment register set with fresh loads (no register rotation copies).
#define LOADSET(F, PA, PB)                    \
  F##a  = ldfrag(PA);                         \
  F##b0 = ldfrag32(PB);                       \
  F##b1 = ldfrag32((PB) + 512);               \
  F##b2 = ldfrag32((PB) + 1024);              \
  F##b3 = ldfrag32((PB) + 1536);

#define WMMA4(F)                              \
  acc0 = WMMA_BF16(F##a, F##b0, acc0);        \
  acc1 = WMMA_BF16(F##a, F##b1, acc1);        \
  acc2 = WMMA_BF16(F##a, F##b2, acc2);        \
  acc3 = WMMA_BF16(F##a, F##b3, acc3);

// ---------------------------------------------------------------------------
// Kernel 0: zero the guard regions (xn row -1, K guard window per head,
// V guard columns per (head,d) row). Re-run every launch: graph-replay safe.
// ---------------------------------------------------------------------------
__global__ void k_init_guards(bf16_t* __restrict__ xg, bf16_t* __restrict__ kg,
                              bf16_t* __restrict__ vt) {
  int tid = blockIdx.x * blockDim.x + threadIdx.x;
  int stride = gridDim.x * blockDim.x;
  for (int i = tid; i < DIM; i += stride) xg[i] = (bf16_t)0.0f;
  const int kguard = HEADS * WS * DH;
  for (int i = tid; i < kguard; i += stride) {
    int head = i / (WS * DH);
    int rest = i - head * (WS * DH);
    kg[(size_t)head * KSTRIDE + rest] = (bf16_t)0.0f;
  }
  const int vguard = HEADS * DH * WS;
  for (int i = tid; i < vguard; i += stride) {
    int row = i / WS;                   // head*DH + d
    int off = i - row * WS;
    vt[(size_t)row * VROW + off] = (bf16_t)0.0f;
  }
}

// ---------------------------------------------------------------------------
// Kernel 1: pack fp32 weight matrix (K x N row-major) into bf16 WMMA
// B-fragment order: dst[((kt*N + n)*2 + h)*16 + e] = W[(kt*32 + koff)*N + n],
// koff = h*8 + (e < 8 ? e : e + 8).
// ---------------------------------------------------------------------------
__global__ void k_pack_b(const float* __restrict__ W, bf16_t* __restrict__ dst,
                         int K, int Ncols) {
  int total = K * Ncols;
  int i = blockIdx.x * blockDim.x + threadIdx.x;
  int stride = gridDim.x * blockDim.x;
  for (; i < total; i += stride) {
    int e    = i & 15;
    int h    = (i >> 4) & 1;
    int rest = i >> 5;
    int n    = rest % Ncols;
    int kt   = rest / Ncols;
    int k    = kt * 32 + h * 8 + (e < 8 ? e : e + 8);
    dst[i] = (bf16_t)W[(size_t)k * Ncols + n];
  }
}

// ---------------------------------------------------------------------------
// Kernel 2: LayerNorm, one row (1024 f32) per 256-thread block, bf16 out
// ---------------------------------------------------------------------------
__global__ void __launch_bounds__(256) k_layernorm(const float* __restrict__ x,
                                                   const float* __restrict__ gamma,
                                                   const float* __restrict__ beta,
                                                   bf16_t* __restrict__ xn) {
  int row = blockIdx.x;
  int tid = threadIdx.x;
  const float4 v = ((const float4*)(x + (size_t)row * DIM))[tid];
  float s  = v.x + v.y + v.z + v.w;
  float ss = v.x * v.x + v.y * v.y + v.z * v.z + v.w * v.w;
#pragma unroll
  for (int off = 16; off > 0; off >>= 1) {
    s  += __shfl_xor(s, off, 32);
    ss += __shfl_xor(ss, off, 32);
  }
  __shared__ float red_s[8], red_ss[8];
  int w = tid >> 5, lane = tid & 31;
  if (lane == 0) { red_s[w] = s; red_ss[w] = ss; }
  __syncthreads();
  if (tid == 0) {
    float ts = 0.f, tss = 0.f;
#pragma unroll
    for (int i = 0; i < 8; ++i) { ts += red_s[i]; tss += red_ss[i]; }
    float mu  = ts / (float)DIM;
    float var = tss / (float)DIM - mu * mu;
    red_s[0]  = mu;
    red_ss[0] = rsqrtf(var + LN_EPS);
  }
  __syncthreads();
  float mu = red_s[0], inv = red_ss[0];
  bf16_t* xo = xn + (size_t)row * DIM;
  int c = tid * 4;
  xo[c + 0] = (bf16_t)((v.x - mu) * inv * gamma[c + 0] + beta[c + 0]);
  xo[c + 1] = (bf16_t)((v.y - mu) * inv * gamma[c + 1] + beta[c + 1]);
  xo[c + 2] = (bf16_t)((v.z - mu) * inv * gamma[c + 2] + beta[c + 2]);
  xo[c + 3] = (bf16_t)((v.w - mu) * inv * gamma[c + 3] + beta[c + 3]);
}

// ---------------------------------------------------------------------------
// Kernel 3: QKV GEMM (token-shift fused on A, rotary fused in epilogue).
// Block = 8 waves, block tile 128(M) x 64(N); wave tile 16 x 64.
// Copy-free ping-pong k-loop; per-lane base pointers hoisted; the token-shift
// is handled by splitting the k-loop into two halves with different A bases.
// ---------------------------------------------------------------------------
__global__ void __launch_bounds__(256) k_qkv_gemm(const bf16_t* __restrict__ xn,
                                                  const bf16_t* __restrict__ wq,
                                                  const float* __restrict__ sinb,
                                                  const float* __restrict__ cosb,
                                                  bf16_t* __restrict__ Qb,
                                                  bf16_t* __restrict__ Kb,
                                                  bf16_t* __restrict__ Vt) {
  __shared__ float Ct[8][16][64];
  int lane = threadIdx.x & 31;
  int w    = threadIdx.x >> 5;
  int n0   = blockIdx.x * 64;
  int m0   = blockIdx.y * 128 + w * 16;
  int cl   = lane & 15;
  int h    = lane >> 4;
  int lo8  = h << 3;

  int n = m0 + cl;
  // A bases: k<512 reads token n-1 (guard row at -1), k>=512 reads token n.
  const bf16_t* paS = xn + ((long long)n - 1) * DIM + lo8;         // + k, k in [0,512)
  const bf16_t* paU = xn + (long long)n * DIM + 512 + lo8;         // + (k-512)
  const size_t  BSTEP = (size_t)QKV_N * 32;                        // per k-tile
  const bf16_t* pb0 = wq + ((size_t)(n0 + cl) * 2 + h) * 16;

  v8f acc0 = {}, acc1 = {}, acc2 = {}, acc3 = {};
  v16bf fa, fb0, fb1, fb2, fb3;
  v16bf ga, gb0, gb1, gb2, gb3;

#pragma unroll 1
  for (int half = 0; half < 2; ++half) {
    const bf16_t* pa = half ? paU : paS;
    const bf16_t* pb = pb0 + (size_t)(half * 16) * BSTEP;
    __builtin_prefetch((const void*)(pb + 16 * BSTEP), 0, 1);
    LOADSET(f, pa, pb)
    int c = 0;
#pragma unroll 1
    for (; c < 512 - 64; c += 64) {
      LOADSET(g, pa + c + 32, pb + (size_t)((c + 32) >> 5) * BSTEP)
      WMMA4(f)
      LOADSET(f, pa + c + 64, pb + (size_t)((c + 64) >> 5) * BSTEP)
      WMMA4(g)
    }
    LOADSET(g, pa + c + 32, pb + (size_t)((c + 32) >> 5) * BSTEP)
    WMMA4(f)
    WMMA4(g)
  }

  // Stage C tiles to LDS in natural (row, col) order.
  int rb = h << 3;
#pragma unroll
  for (int vi = 0; vi < 8; ++vi) {
    Ct[w][rb + vi][cl]      = acc0[vi];
    Ct[w][rb + vi][16 + cl] = acc1[vi];
    Ct[w][rb + vi][32 + cl] = acc2[vi];
    Ct[w][rb + vi][48 + cl] = acc3[vi];
  }
  __syncthreads();

  // This 64-wide tile is entirely inside one of q/k/v and one head.
  int part = n0 >> 9;             // 0=q, 1=k, 2=v
  int head = (n0 & 511) >> 6;

  // Rotary over (even, odd) pairs: 512 pairs / 32 lanes = 16 per lane.
  if (part < 2) {
    bf16_t* dst = (part == 0) ? (Qb + (size_t)head * N_TOK * DH)
                              : (Kb + (size_t)head * KSTRIDE + (size_t)WS * DH);
#pragma unroll
    for (int t = 0; t < 16; ++t) {
      int p  = lane + 32 * t;
      int r  = p >> 5;
      int cp = (p & 31) * 2;      // dh index of the even element
      int tok = m0 + r;
      float t0 = Ct[w][r][cp], t1 = Ct[w][r][cp + 1];
      float c0 = cosb[(size_t)tok * DH + cp];
      float c1 = cosb[(size_t)tok * DH + cp + 1];
      float s0 = sinb[(size_t)tok * DH + cp];
      float s1 = sinb[(size_t)tok * DH + cp + 1];
      v8bf pk2 = {};              // pack pair into one 4-byte store
      pk2[0] = (bf16_t)(t0 * c0 - t1 * s0);
      pk2[1] = (bf16_t)(t1 * c1 + t0 * s1);
      *(uint*)(dst + (size_t)tok * DH + cp) = *(uint*)&pk2;
    }
  } else {
    bf16_t* dst = Vt + (size_t)head * DH * VROW + WS;
#pragma unroll
    for (int t = 0; t < 16; ++t) {
      int p  = lane + 32 * t;
      int r  = p >> 5;
      int cp = (p & 31) * 2;
      int tok = m0 + r;
      float t0 = Ct[w][r][cp], t1 = Ct[w][r][cp + 1];
      float c0 = cosb[(size_t)tok * DH + cp];
      float c1 = cosb[(size_t)tok * DH + cp + 1];
      float s0 = sinb[(size_t)tok * DH + cp];
      float s1 = sinb[(size_t)tok * DH + cp + 1];
      dst[(size_t)cp * VROW + tok]       = (bf16_t)(t0 * c0 - t1 * s0);
      dst[(size_t)(cp + 1) * VROW + tok] = (bf16_t)(t1 * c1 + t0 * s1);
    }
  }
}

// ---------------------------------------------------------------------------
// Kernel 4: windowed attention with one-window lookback, flash-style online
// softmax. One block per (head, window, half); wave owns 16 query rows.
// Guard windows make all K/V loads branch-free; per-lane base pointers are
// hoisted so inner-loop addresses are simple increments.
// ---------------------------------------------------------------------------
__global__ void __launch_bounds__(256) k_attention(const bf16_t* __restrict__ Qb,
                                                   const bf16_t* __restrict__ Kb,
                                                   const bf16_t* __restrict__ Vt,
                                                   bf16_t* __restrict__ attn_out) {
  __shared__ float  Sbuf[8][16][32];
  __shared__ __align__(16) bf16_t Pbuf[8][16][32];
  __shared__ float  mrow[8][16], lrow[8][16], crow[8][16];

  int lane = threadIdx.x & 31;
  int w    = threadIdx.x >> 5;
  int bid  = blockIdx.x;                 // head*256 + win*2 + half
  int half = bid & 1;
  int win  = (bid >> 1) & (NWIN - 1);
  int head = bid >> 8;
  int cl   = lane & 15;
  int h    = lane >> 4;
  int lo8  = h << 3;

  const bf16_t* Qh  = Qb + (size_t)head * N_TOK * DH;
  const bf16_t* Kh  = Kb + (size_t)head * KSTRIDE + (size_t)WS * DH;
  const bf16_t* Vth = Vt + (size_t)head * DH * VROW + WS;

  int irow0_in    = half * 128 + w * 16;                // row-in-window base
  long long qrow0 = (long long)win * WS + irow0_in;     // global token base
  long long krow0 = (long long)(win - 1) * WS;          // lookback start (>= -WS)

  // Per-lane hoisted bases.
  const bf16_t* qbase = Qh + (qrow0 + cl) * DH + lo8;
  const bf16_t* kbase = Kh + (krow0 + cl) * DH + lo8;   // + j*DH per column
  const bf16_t* vbase = Vth + (size_t)cl * VROW + krow0 + lo8;   // + 16*VROW per d-tile

  v16bf qa0 = ldfrag(qbase);
  v16bf qa1 = ldfrag(qbase + 32);
  v8f o0 = {}, o1 = {}, o2 = {}, o3 = {};

  if (lane < 16) { mrow[w][lane] = -3.0e38f; lrow[w][lane] = 0.0f; }
  __syncthreads();

  int rb = h << 3;

  for (int jc = 0; jc < 16; ++jc) {       // 512 / 32 key chunks
    int j0 = jc * 32;
#pragma unroll
    for (int jt = 0; jt < 2; ++jt) {
      v8f s = {};
      const bf16_t* kp = kbase + (size_t)(j0 + jt * 16) * DH;
      v16bf kf0 = ldfrag(kp);
      v16bf kf1 = ldfrag(kp + 32);
      s = WMMA_BF16(qa0, kf0, s);
      s = WMMA_BF16(qa1, kf1, s);
#pragma unroll
      for (int vi = 0; vi < 8; ++vi) {
        int r = rb + vi;
        int j = j0 + jt * 16 + cl;
        float val = s[vi] * 0.125f;                    // DH^-0.5
        if (j > irow0_in + r + WS) val = MASK_VALF;    // tril(.., k=WS)
        Sbuf[w][r][jt * 16 + cl] = val;
      }
    }
    __syncthreads();

    // Online softmax: partner lanes (lane, lane^16) share row r = lane&15;
    // each handles 16 columns.
    {
      int r  = cl;
      int ch = h << 4;                    // 0 or 16
      float mold = mrow[w][r];
      float tmax = -3.0e38f;
#pragma unroll
      for (int c = 0; c < 16; ++c) tmax = fmaxf(tmax, Sbuf[w][r][ch + c]);
      tmax = fmaxf(tmax, __shfl_xor(tmax, 16, 32));
      tmax = fmaxf(tmax, mold);
      float sum = 0.f;
#pragma unroll
      for (int c = 0; c < 16; ++c) {
        float p = __expf(Sbuf[w][r][ch + c] - tmax);
        sum += p;
        Pbuf[w][r][ch + c] = (bf16_t)p;
      }
      sum += __shfl_xor(sum, 16, 32);
      if (lane < 16) {
        float corr = __expf(mold - tmax);
        mrow[w][r] = tmax;
        lrow[w][r] = lrow[w][r] * corr + sum;
        crow[w][r] = corr;
      }
    }
    __syncthreads();

    // Rescale accumulators, then O += P @ V.
#pragma unroll
    for (int vi = 0; vi < 8; ++vi) {
      float f = crow[w][rb + vi];
      o0[vi] *= f; o1[vi] *= f; o2[vi] *= f; o3[vi] *= f;
    }
    v16bf pa;
    {
      const v8bf* pp = (const v8bf*)(&Pbuf[w][cl][lo8]);   // 16B aligned
      pa = combine(pp[0], pp[2]);                          // +0 and +16 elems
    }
    const bf16_t* vp = vbase + j0;
    v16bf vf0 = combine(*(const v8bf*)vp, *(const v8bf*)(vp + 16));
    vp += (size_t)16 * VROW;
    v16bf vf1 = combine(*(const v8bf*)vp, *(const v8bf*)(vp + 16));
    vp += (size_t)16 * VROW;
    v16bf vf2 = combine(*(const v8bf*)vp, *(const v8bf*)(vp + 16));
    vp += (size_t)16 * VROW;
    v16bf vf3 = combine(*(const v8bf*)vp, *(const v8bf*)(vp + 16));
    o0 = WMMA_BF16(pa, vf0, o0);
    o1 = WMMA_BF16(pa, vf1, o1);
    o2 = WMMA_BF16(pa, vf2, o2);
    o3 = WMMA_BF16(pa, vf3, o3);
    __syncthreads();
  }

  // Finalize: divide by row sums, write bf16 to [token][head*64 + d].
#pragma unroll
  for (int vi = 0; vi < 8; ++vi) {
    int r = rb + vi;
    float inv = 1.0f / lrow[w][r];
    long long tok = qrow0 + r;
    bf16_t* dst = attn_out + tok * INNER + head * DH;
    dst[cl]      = (bf16_t)(o0[vi] * inv);
    dst[16 + cl] = (bf16_t)(o1[vi] * inv);
    dst[32 + cl] = (bf16_t)(o2[vi] * inv);
    dst[48 + cl] = (bf16_t)(o3[vi] * inv);
  }
}

// ---------------------------------------------------------------------------
// Kernel 5: output GEMM (N x 512) @ (512 x 1024) + bias -> fp32,
// copy-free ping-pong k-loop.
// ---------------------------------------------------------------------------
__global__ void __launch_bounds__(256) k_out_gemm(const bf16_t* __restrict__ ao,
                                                  const bf16_t* __restrict__ wo,
                                                  const float* __restrict__ bo,
                                                  float* __restrict__ out) {
  int lane = threadIdx.x & 31;
  int w    = threadIdx.x >> 5;
  int n0   = blockIdx.x * 64;
  int m0   = blockIdx.y * 128 + w * 16;
  int cl   = lane & 15;
  int h    = lane >> 4;
  int lo8  = h << 3;

  const bf16_t* pa = ao + (size_t)(m0 + cl) * INNER + lo8;
  const size_t  BSTEP = (size_t)DIM * 32;
  const bf16_t* pb = wo + ((size_t)(n0 + cl) * 2 + h) * 16;

  v8f acc0 = {}, acc1 = {}, acc2 = {}, acc3 = {};
  v16bf fa, fb0, fb1, fb2, fb3;
  v16bf ga, gb0, gb1, gb2, gb3;

  __builtin_prefetch((const void*)(pb + 8 * BSTEP), 0, 1);
  LOADSET(f, pa, pb)
  int c = 0;
#pragma unroll 1
  for (; c < INNER - 64; c += 64) {
    LOADSET(g, pa + c + 32, pb + (size_t)((c + 32) >> 5) * BSTEP)
    WMMA4(f)
    LOADSET(f, pa + c + 64, pb + (size_t)((c + 64) >> 5) * BSTEP)
    WMMA4(g)
  }
  LOADSET(g, pa + c + 32, pb + (size_t)((c + 32) >> 5) * BSTEP)
  WMMA4(f)
  WMMA4(g)

  int rb = h << 3;
#pragma unroll
  for (int vi = 0; vi < 8; ++vi) {
    int row = m0 + rb + vi;
    float* orow = out + (size_t)row * DIM + n0;
    orow[cl]      = acc0[vi] + bo[n0 + cl];
    orow[16 + cl] = acc1[vi] + bo[n0 + 16 + cl];
    orow[32 + cl] = acc2[vi] + bo[n0 + 32 + cl];
    orow[48 + cl] = acc3[vi] + bo[n0 + 48 + cl];
  }
}

// ---------------------------------------------------------------------------
// Launch
// ---------------------------------------------------------------------------
extern "C" void kernel_launch(void* const* d_in, const int* in_sizes, int n_in,
                              void* d_out, int out_size, void* d_ws, size_t ws_size,
                              hipStream_t stream) {
  const float* x     = (const float*)d_in[0];
  const float* sinb  = (const float*)d_in[1];
  const float* cosb  = (const float*)d_in[2];
  const float* gamma = (const float*)d_in[3];
  const float* beta  = (const float*)d_in[4];
  const float* w_qkv = (const float*)d_in[5];
  const float* w_out = (const float*)d_in[6];
  const float* b_out = (const float*)d_in[7];
  float* out = (float*)d_out;

  char* ws = (char*)d_ws;
  size_t off = 0;
  auto alloc = [&](size_t bytes) -> char* {
    char* p = ws + off;
    off += (bytes + 255) & ~(size_t)255;
    return p;
  };

  bf16_t* xn_base = (bf16_t*)alloc((size_t)(N_TOK + 1) * DIM * sizeof(bf16_t));
  bf16_t* wq_pk   = (bf16_t*)alloc((size_t)DIM * QKV_N * sizeof(bf16_t));
  bf16_t* wo_pk   = (bf16_t*)alloc((size_t)INNER * DIM * sizeof(bf16_t));
  bf16_t* Qb      = (bf16_t*)alloc((size_t)HEADS * N_TOK * DH * sizeof(bf16_t));
  bf16_t* Kb      = (bf16_t*)alloc((size_t)HEADS * KSTRIDE * sizeof(bf16_t));
  bf16_t* Vt      = (bf16_t*)alloc((size_t)HEADS * DH * VROW * sizeof(bf16_t));
  bf16_t* attn    = (bf16_t*)alloc((size_t)N_TOK * INNER * sizeof(bf16_t));

  bf16_t* xn = xn_base + DIM;           // row -1 = zero guard row

  k_init_guards<<<256, 256, 0, stream>>>(xn_base, Kb, Vt);
  k_pack_b<<<512, 256, 0, stream>>>(w_qkv, wq_pk, DIM, QKV_N);
  k_pack_b<<<512, 256, 0, stream>>>(w_out, wo_pk, INNER, DIM);
  k_layernorm<<<N_TOK, 256, 0, stream>>>(x, gamma, beta, xn);
  k_qkv_gemm<<<dim3(QKV_N / 64, N_TOK / 128), 256, 0, stream>>>(xn, wq_pk, sinb, cosb, Qb, Kb, Vt);
  k_attention<<<HEADS * NWIN * 2, 256, 0, stream>>>(Qb, Kb, Vt, attn);
  k_out_gemm<<<dim3(DIM / 64, N_TOK / 128), 256, 0, stream>>>(attn, wo_pk, b_out, out);
}
